// Block_45715631899385
// MI455X (gfx1250) — matrix operand order repaired and checked by reading, not compiled, not run on hardware
//
#include <hip/hip_runtime.h>
#include <hip/hip_bf16.h>
#include <math.h>

// ---------------------------------------------------------------------------
// Transformer block fwd on MI455X (gfx1250): bf16 WMMA everywhere.
//   D=512, H=8, dq=64, T=2048, B=4, MLP=2048.
// ---------------------------------------------------------------------------

#define BATCH 4
#define TSEQ  2048
#define DMODEL 512
#define HEADS 8
#define DQ    64
#define MLPD  2048
#define NTOK  (BATCH * TSEQ)   // 8192

typedef __bf16 bf16;
typedef __attribute__((ext_vector_type(16))) __bf16 v16bf;
typedef __attribute__((ext_vector_type(8)))  float  v8f;
typedef __attribute__((ext_vector_type(4)))  int    v4i;

union BF16Frag { v16bf v; unsigned int u[8]; bf16 h[16]; };
union AccFrag  { v8f v; float f[8]; };
union BFU      { unsigned short s; bf16 b; };
union U4BF     { uint4 q; bf16 h[8]; };

#if __has_builtin(__builtin_amdgcn_global_load_async_to_lds_b128)
#define HAVE_ASYNC_LDS 1
typedef __attribute__((address_space(1))) v4i GlobV4;
typedef __attribute__((address_space(3))) v4i LdsV4;
#else
#define HAVE_ASYNC_LDS 0
#endif

// round-to-nearest-even fp32 -> bf16 via bits (no reliance on cast codegen)
__device__ __forceinline__ bf16 f2bf(float f) {
    unsigned u = __float_as_uint(f);
    BFU r; r.s = (unsigned short)((u + 0x7FFFu + ((u >> 16) & 1u)) >> 16);
    return r.b;
}

__device__ __forceinline__ v8f wmma_bf16(v16bf a, v16bf b, v8f c) {
    return __builtin_amdgcn_wmma_f32_16x16x32_bf16(
        false, a, false, b, (short)0, c, false, false);
}

// ---------------------------------------------------------------------------
// fp32 -> bf16 convert (weights)
// ---------------------------------------------------------------------------
__global__ void cvt_f32_bf16(const float* __restrict__ src,
                             bf16* __restrict__ dst, int n) {
    int i = blockIdx.x * blockDim.x + threadIdx.x;
    if (i < n) dst[i] = f2bf(src[i]);
}

// ---------------------------------------------------------------------------
// LayerNorm over last dim (512), one block per row, fp32 in -> bf16 out
// ---------------------------------------------------------------------------
__global__ __launch_bounds__(256)
void ln_bf16_kernel(const float* __restrict__ x, const float* __restrict__ g,
                    const float* __restrict__ b, bf16* __restrict__ out) {
    __shared__ float red[256];
    int row = blockIdx.x;
    int tid = threadIdx.x;
    const float* xr = x + (size_t)row * DMODEL;
    float a0 = xr[tid], a1 = xr[tid + 256];

    red[tid] = a0 + a1;
    __syncthreads();
    for (int off = 128; off > 0; off >>= 1) {
        if (tid < off) red[tid] += red[tid + off];
        __syncthreads();
    }
    float mu = red[0] * (1.0f / DMODEL);
    __syncthreads();

    float d0 = a0 - mu, d1 = a1 - mu;
    red[tid] = d0 * d0 + d1 * d1;
    __syncthreads();
    for (int off = 128; off > 0; off >>= 1) {
        if (tid < off) red[tid] += red[tid + off];
        __syncthreads();
    }
    float rstd = rsqrtf(red[0] * (1.0f / DMODEL) + 1e-5f);

    bf16* orow = out + (size_t)row * DMODEL;
    orow[tid]       = f2bf(d0 * rstd * g[tid]       + b[tid]);
    orow[tid + 256] = f2bf(d1 * rstd * g[tid + 256] + b[tid + 256]);
}

// ---------------------------------------------------------------------------
// bf16 GEMM: C[M,N] = epi(A[M,K] * B[K,N] + bias [+ res])
//   block: 128 threads = 4 waves; block tile 128x64; wave tile 32x64
//   K staged in 32-wide chunks through LDS (stride 40 -> conflict free, 16B ok)
//   EPI: 0 = bf16 out, 1 = exact-GELU bf16 out, 2 = fp32 out with residual,
//        3 = bf16 out stored transposed per batch: Out[b][col][t]  (for V)
// ---------------------------------------------------------------------------
#define LDSTRIDE 40

template <int EPI>
__global__ __launch_bounds__(128)
void gemm_bf16(const bf16* __restrict__ A, const bf16* __restrict__ Bm,
               const float* __restrict__ bias, const float* __restrict__ res,
               void* __restrict__ Out, int M, int N, int K) {
    __shared__ bf16 sA[128 * LDSTRIDE];
    __shared__ bf16 sB[64 * LDSTRIDE];   // transposed: [n][k]

    const int tid  = threadIdx.x;
    const int wave = tid >> 5;
    const int lane = tid & 31;
    const int m16  = lane & 15;
    const int half = lane >> 4;

    const size_t rowBase = (size_t)blockIdx.x * 128;
    const int    colBase = blockIdx.y * 64;

    AccFrag acc[2][4];
#pragma unroll
    for (int s = 0; s < 2; ++s)
#pragma unroll
        for (int nb = 0; nb < 4; ++nb)
#pragma unroll
            for (int i = 0; i < 8; ++i) acc[s][nb].f[i] = 0.0f;

    for (int kc = 0; kc < K; kc += 32) {
        // --- stage A tile: thread tid copies its row, 32 bf16 = 4 x b128 ---
        {
            const bf16* srow = A + (rowBase + tid) * (size_t)K + kc;
#if HAVE_ASYNC_LDS
            // gfx1250 async global->LDS DMA path (ASYNCcnt tracked)
#pragma unroll
            for (int j = 0; j < 4; ++j) {
                __builtin_amdgcn_global_load_async_to_lds_b128(
                    (GlobV4*)(srow + j * 8),
                    (LdsV4*)(&sA[tid * LDSTRIDE + j * 8]),
                    0, 0);
            }
#else
            const uint4* src = (const uint4*)srow;
            uint4* dst = (uint4*)(&sA[tid * LDSTRIDE]);
#pragma unroll
            for (int j = 0; j < 4; ++j) dst[j] = src[j];
#endif
        }
        // --- stage B tile transposed: kk = tid>>2 (0..31), qd = tid&3 ---
        {
            const int kk = tid >> 2, qd = tid & 3;
            const bf16* src = Bm + (size_t)(kc + kk) * N + colBase + qd * 16;
            U4BF t0, t1;
            t0.q = ((const uint4*)src)[0];
            t1.q = ((const uint4*)src)[1];
#pragma unroll
            for (int j = 0; j < 8; ++j) {
                sB[(qd * 16 + j)     * LDSTRIDE + kk] = t0.h[j];
                sB[(qd * 16 + j + 8) * LDSTRIDE + kk] = t1.h[j];
            }
        }
#if HAVE_ASYNC_LDS
        asm volatile("s_wait_asynccnt 0x0" ::: "memory");
#endif
        __syncthreads();

        // --- per-lane operand fragments (ISA 16-bit A / B wave32 layouts) ---
        BF16Frag af[2], bfr[4];
#pragma unroll
        for (int s = 0; s < 2; ++s) {
            const int m = wave * 32 + s * 16 + m16;
#pragma unroll
            for (int v = 0; v < 8; ++v) {
                const int k0 = (v < 4) ? (8 * half + 2 * v)
                                       : (16 + 8 * half + 2 * (v - 4));
                af[s].u[v] = *(const unsigned int*)&sA[m * LDSTRIDE + k0];
            }
        }
#pragma unroll
        for (int nb = 0; nb < 4; ++nb) {
            const int n = nb * 16 + m16;
#pragma unroll
            for (int v = 0; v < 8; ++v) {
                const int k0 = (v < 4) ? (8 * half + 2 * v) : (16 + 8 * half + 2 * (v - 4));
                bfr[nb].u[v] = *(const unsigned int*)&sB[n * LDSTRIDE + k0];
            }
        }
#pragma unroll
        for (int s = 0; s < 2; ++s)
#pragma unroll
            for (int nb = 0; nb < 4; ++nb)
                acc[s][nb].v = wmma_bf16(af[s].v, bfr[nb].v, acc[s][nb].v);
        __syncthreads();
    }

    // --- epilogue: D layout = vgpr i -> row i + 8*half, col = lane%16 ---
#pragma unroll
    for (int s = 0; s < 2; ++s)
#pragma unroll
        for (int nb = 0; nb < 4; ++nb)
#pragma unroll
            for (int i = 0; i < 8; ++i) {
                const size_t row = rowBase + wave * 32 + s * 16 + i + 8 * half;
                const int    col = colBase + nb * 16 + m16;
                float vv = acc[s][nb].f[i] + bias[col];
                if (EPI == 1) {  // exact GELU
                    vv = 0.5f * vv * (1.0f + erff(vv * 0.70710678118654752f));
                }
                if (EPI == 2) {
                    vv += res[row * (size_t)N + col];
                    ((float*)Out)[row * (size_t)N + col] = vv;
                } else if (EPI == 3) {
                    const size_t bb = row / TSEQ, t = row % TSEQ;
                    ((bf16*)Out)[(bb * N + col) * (size_t)TSEQ + t] = f2bf(vv);
                } else {
                    ((bf16*)Out)[row * (size_t)N + col] = f2bf(vv);
                }
            }
}

// ---------------------------------------------------------------------------
// Flash attention: one wave per (qtile16, head, batch).
//   Q/K are bf16 [B][T][D] (head h = cols h*64..h*64+63); V is pre-transposed
//   bf16 [B][D][T] so every P*V B-fragment register is one 32-bit load.
//   32 keys per step: 4 WMMA for S (dq=64 contraction), online softmax via
//   LDS (also converts D-layout S -> A-layout P), 4 WMMA for P*V.
// ---------------------------------------------------------------------------
__global__ __launch_bounds__(32)
void attn_kernel(const bf16* __restrict__ Q, const bf16* __restrict__ Kmat,
                 const bf16* __restrict__ Vt, bf16* __restrict__ Oa) {
    const int qt = blockIdx.x, h = blockIdx.y, b = blockIdx.z;
    const int lane = threadIdx.x;
    const int n16  = lane & 15;
    const int half = lane >> 4;

    const size_t bo = (size_t)b * TSEQ * DMODEL + (size_t)h * DQ;
    const bf16* Qb  = Q    + bo;
    const bf16* Kb  = Kmat + bo;
    const bf16* Vtb = Vt + ((size_t)b * DMODEL + (size_t)h * DQ) * TSEQ;
    const int q0 = qt * 16;

    __shared__ float sS[16][36];
    __shared__ bf16  sP[16][36];
    __shared__ float sM[16], sL[16], sF[16];

    // Q A-fragments: qa[g] covers contraction d in [g*32, g*32+32)
    BF16Frag qa[2];
    const int qrow = q0 + n16;
#pragma unroll
    for (int g = 0; g < 2; ++g)
#pragma unroll
        for (int v = 0; v < 8; ++v) {
            const int k0 = g * 32 + ((v < 4) ? (8 * half + 2 * v)
                                             : (16 + 8 * half + 2 * (v - 4)));
            qa[g].u[v] = *(const unsigned int*)(Qb + (size_t)qrow * DMODEL + k0);
        }

    if (lane < 16) { sM[lane] = -1e30f; sL[lane] = 0.0f; }
    AccFrag oacc[4];
#pragma unroll
    for (int nb = 0; nb < 4; ++nb)
#pragma unroll
        for (int i = 0; i < 8; ++i) oacc[nb].f[i] = 0.0f;
    __syncthreads();

    for (int kb = 0; kb < TSEQ; kb += 32) {
        // prefetch next K/V tiles (global_prefetch_b8)
        if (kb + 32 < TSEQ) {
            __builtin_prefetch(Kb + (size_t)(kb + 32 + lane) * DMODEL, 0, 1);
            __builtin_prefetch(Vtb + (size_t)(lane * 2) * TSEQ + kb + 32, 0, 1);
        }

        // ---- S = Q * K^T for 2 x 16 keys, scaled, into LDS ----
#pragma unroll
        for (int st = 0; st < 2; ++st) {
            const int kbase = kb + st * 16;
            BF16Frag kbf[2];
#pragma unroll
            for (int g = 0; g < 2; ++g)
#pragma unroll
                for (int v = 0; v < 8; ++v) {
                    const int dk = g * 32 + ((v < 4) ? (8 * half + 2 * v) : (16 + 8 * half + 2 * (v - 4)));
                    kbf[g].u[v] = *(const unsigned int*)
                        (Kb + (size_t)(kbase + n16) * DMODEL + dk);
                }
            AccFrag s;
#pragma unroll
            for (int i = 0; i < 8; ++i) s.f[i] = 0.0f;
            s.v = wmma_bf16(qa[0].v, kbf[0].v, s.v);
            s.v = wmma_bf16(qa[1].v, kbf[1].v, s.v);
#pragma unroll
            for (int i = 0; i < 8; ++i)
                sS[i + 8 * half][st * 16 + n16] = s.f[i] * 0.125f;  // dq^-0.5
        }
        __syncthreads();

        // ---- online softmax: lanes 0..15 own one query row each ----
        if (lane < 16) {
            const int r = lane;
            float sv[32];
            float tm = -1e30f;
#pragma unroll
            for (int j = 0; j < 32; ++j) { sv[j] = sS[r][j]; tm = fmaxf(tm, sv[j]); }
            const float mold = sM[r];
            const float mnew = fmaxf(mold, tm);
            const float fsc  = __expf(mold - mnew);
            float lsum = 0.0f;
#pragma unroll
            for (int j = 0; j < 32; ++j) {
                const float p = __expf(sv[j] - mnew);
                sP[r][j] = f2bf(p);
                lsum += p;
            }
            sL[r] = sL[r] * fsc + lsum;
            sM[r] = mnew;
            sF[r] = fsc;
        }
        __syncthreads();

        // ---- rescale O accumulators by per-row factor ----
        float fr[8];
#pragma unroll
        for (int i = 0; i < 8; ++i) fr[i] = sF[i + 8 * half];
#pragma unroll
        for (int nb = 0; nb < 4; ++nb)
#pragma unroll
            for (int i = 0; i < 8; ++i) oacc[nb].f[i] *= fr[i];

        // ---- P as A-fragment (rows = queries, contraction = 32 keys) ----
        BF16Frag pa;
#pragma unroll
        for (int v = 0; v < 8; ++v) {
            const int k0 = (v < 4) ? (8 * half + 2 * v)
                                   : (16 + 8 * half + 2 * (v - 4));
            pa.u[v] = *(const unsigned int*)&sP[n16][k0];
        }

        // ---- O += P * V : 4 n-tiles over dq; V^T makes each reg 1 load ----
#pragma unroll
        for (int nb = 0; nb < 4; ++nb) {
            BF16Frag vbf;
#pragma unroll
            for (int v = 0; v < 8; ++v) {
                const int kt0 = (v < 4) ? (8 * half + 2 * v) : (16 + 8 * half + 2 * (v - 4));
                vbf.u[v] = *(const unsigned int*)
                    (Vtb + (size_t)(nb * 16 + n16) * TSEQ + kb + kt0);
            }
            oacc[nb].v = wmma_bf16(pa.v, vbf.v, oacc[nb].v);
        }
        __syncthreads();   // guard sS/sP/sF reuse next iteration
    }

    // ---- finalize: divide by l, store bf16 [b][t][h*64+col] ----
    float lr[8];
#pragma unroll
    for (int i = 0; i < 8; ++i) lr[i] = 1.0f / sL[i + 8 * half];
#pragma unroll
    for (int nb = 0; nb < 4; ++nb)
#pragma unroll
        for (int i = 0; i < 8; ++i) {
            const int t = q0 + i + 8 * half;
            Oa[bo + (size_t)t * DMODEL + nb * 16 + n16] = f2bf(oacc[nb].f[i] * lr[i]);
        }
}

// ---------------------------------------------------------------------------
// Host-side launch
// ---------------------------------------------------------------------------
extern "C" void kernel_launch(void* const* d_in, const int* in_sizes, int n_in,
                              void* d_out, int out_size, void* d_ws, size_t ws_size,
                              hipStream_t stream) {
    const float* x     = (const float*)d_in[0];
    const float* ln1_g = (const float*)d_in[1];
    const float* ln1_b = (const float*)d_in[2];
    const float* Wq    = (const float*)d_in[3];
    const float* bq    = (const float*)d_in[4];
    const float* Wk    = (const float*)d_in[5];
    const float* bk    = (const float*)d_in[6];
    const float* Wv    = (const float*)d_in[7];
    const float* bv    = (const float*)d_in[8];
    const float* Wo    = (const float*)d_in[9];
    const float* bo    = (const float*)d_in[10];
    const float* ln2_g = (const float*)d_in[11];
    const float* ln2_b = (const float*)d_in[12];
    const float* W1    = (const float*)d_in[13];
    const float* b1    = (const float*)d_in[14];
    const float* W2    = (const float*)d_in[15];
    const float* b2    = (const float*)d_in[16];

    // workspace carve-up (256B aligned)
    size_t off = 0;
    auto carve = [&](size_t bytes) -> char* {
        char* p = (char*)d_ws + off;
        off += (bytes + 255) & ~(size_t)255;
        return p;
    };
    bf16*  wq_b = (bf16*)carve((size_t)DMODEL * DMODEL * 2);
    bf16*  wk_b = (bf16*)carve((size_t)DMODEL * DMODEL * 2);
    bf16*  wv_b = (bf16*)carve((size_t)DMODEL * DMODEL * 2);
    bf16*  wo_b = (bf16*)carve((size_t)DMODEL * DMODEL * 2);
    bf16*  w1_b = (bf16*)carve((size_t)DMODEL * MLPD * 2);
    bf16*  w2_b = (bf16*)carve((size_t)MLPD * DMODEL * 2);
    bf16*  xn   = (bf16*)carve((size_t)NTOK * DMODEL * 2);
    bf16*  qb   = (bf16*)carve((size_t)NTOK * DMODEL * 2);
    bf16*  kbuf = (bf16*)carve((size_t)NTOK * DMODEL * 2);
    bf16*  vt   = (bf16*)carve((size_t)NTOK * DMODEL * 2);   // [B][D][T]
    bf16*  attn = (bf16*)carve((size_t)NTOK * DMODEL * 2);
    float* x2   = (float*)carve((size_t)NTOK * DMODEL * 4);
    bf16*  yn   = (bf16*)carve((size_t)NTOK * DMODEL * 2);
    bf16*  hbuf = (bf16*)carve((size_t)NTOK * MLPD * 2);
    (void)ws_size; (void)n_in; (void)in_sizes; (void)out_size;

    // 1) weights -> bf16
    {
        const int ndd = DMODEL * DMODEL, ndm = DMODEL * MLPD;
        cvt_f32_bf16<<<(ndd + 255) / 256, 256, 0, stream>>>(Wq, wq_b, ndd);
        cvt_f32_bf16<<<(ndd + 255) / 256, 256, 0, stream>>>(Wk, wk_b, ndd);
        cvt_f32_bf16<<<(ndd + 255) / 256, 256, 0, stream>>>(Wv, wv_b, ndd);
        cvt_f32_bf16<<<(ndd + 255) / 256, 256, 0, stream>>>(Wo, wo_b, ndd);
        cvt_f32_bf16<<<(ndm + 255) / 256, 256, 0, stream>>>(W1, w1_b, ndm);
        cvt_f32_bf16<<<(ndm + 255) / 256, 256, 0, stream>>>(W2, w2_b, ndm);
    }

    // 2) LN1
    ln_bf16_kernel<<<NTOK, 256, 0, stream>>>(x, ln1_g, ln1_b, xn);

    // 3) Q/K projections (bf16, [b][t][h*64+i]); V projection stored [b][d][t]
    gemm_bf16<0><<<dim3(NTOK / 128, DMODEL / 64), 128, 0, stream>>>(
        xn, wq_b, bq, nullptr, qb, NTOK, DMODEL, DMODEL);
    gemm_bf16<0><<<dim3(NTOK / 128, DMODEL / 64), 128, 0, stream>>>(
        xn, wk_b, bk, nullptr, kbuf, NTOK, DMODEL, DMODEL);
    gemm_bf16<3><<<dim3(NTOK / 128, DMODEL / 64), 128, 0, stream>>>(
        xn, wv_b, bv, nullptr, vt, NTOK, DMODEL, DMODEL);

    // 4) attention (flash, online softmax)
    attn_kernel<<<dim3(TSEQ / 16, HEADS, BATCH), 32, 0, stream>>>(
        qb, kbuf, vt, attn);

    // 5) output projection + residual -> x2 (fp32)
    gemm_bf16<2><<<dim3(NTOK / 128, DMODEL / 64), 128, 0, stream>>>(
        attn, wo_b, bo, x, x2, NTOK, DMODEL, DMODEL);

    // 6) LN2
    ln_bf16_kernel<<<NTOK, 256, 0, stream>>>(x2, ln2_g, ln2_b, yn);

    // 7) MLP up + exact GELU (bf16)
    gemm_bf16<1><<<dim3(NTOK / 128, MLPD / 64), 128, 0, stream>>>(
        yn, w1_b, b1, nullptr, hbuf, NTOK, MLPD, DMODEL);

    // 8) MLP down + residual -> d_out (fp32)
    gemm_bf16<2><<<dim3(NTOK / 128, DMODEL / 64), 128, 0, stream>>>(
        hbuf, w2_b, b2, x2, (float*)d_out, NTOK, DMODEL, MLPD);
}
